// GlobalAttention_6004364280647
// MI455X (gfx1250) — matrix operand order repaired and checked by reading, not compiled, not run on hardware
//
#include <hip/hip_runtime.h>
#include <hip/hip_bf16.h>

// ---------------------------------------------------------------------------
// MI455X (gfx1250) global-attention pipeline, all matmuls via
// v_wmma_f32_16x16x32_bf16 (wave32), async global->LDS staging (ASYNCcnt)
// with double buffering where available.
// ---------------------------------------------------------------------------

typedef __attribute__((ext_vector_type(8)))  float  v8f;
typedef __attribute__((ext_vector_type(8)))  __bf16 v8bf;
typedef __attribute__((ext_vector_type(16))) __bf16 v16bf;
typedef __attribute__((ext_vector_type(4)))  int    v4i;

#define AS1 __attribute__((address_space(1)))
#define AS3 __attribute__((address_space(3)))

#define B_      4
#define C_      256
#define HW_     112
#define P_      12544      // 112*112
#define INNER_  512        // heads*dim_head
#define HEADS_  8
#define D_      64
#define NKV_    256        // 16*16 downsampled K/V tokens
#define SCALE_  0.125f     // 64^-0.5

#ifndef __has_builtin
#define __has_builtin(x) 0
#endif
#if __has_builtin(__builtin_amdgcn_global_load_async_to_lds_b128)
#define ASYNC_COPY 1
#else
#define ASYNC_COPY 0
#endif

// 16B global -> LDS copy; async (VGPR-bypassing, ASYNCcnt) when available.
// Builtin signature (from hipcc diagnostic): (v4i AS1*, v4i AS3*, imm, imm).
static __device__ __forceinline__ void cp16(__bf16* lds, const __bf16* g) {
#if ASYNC_COPY
  __builtin_amdgcn_global_load_async_to_lds_b128(
      (AS1 v4i*)const_cast<__bf16*>(g),
      (AS3 v4i*)lds, 0, 0);
#else
  *(uint4*)lds = *(const uint4*)g;
#endif
}

static __device__ __forceinline__ void wait_async() {
#if ASYNC_COPY
#if __has_builtin(__builtin_amdgcn_s_wait_asynccnt)
  __builtin_amdgcn_s_wait_asynccnt(0);
#else
  asm volatile("s_wait_asynccnt 0" ::: "memory");
#endif
#endif
}

// D = A*B + C, bf16 in / f32 accum
static __device__ __forceinline__ v8f wmma_bf16(v16bf a, v16bf b, v8f c) {
  return __builtin_amdgcn_wmma_f32_16x16x32_bf16(
      /*neg_a*/false, a, /*neg_b*/false, b, /*c_mod*/(short)0, c,
      /*reuse_a*/false, /*reuse_b*/false);
}

// Build a 16-element bf16 fragment from two (possibly non-adjacent) 16B groups
static __device__ __forceinline__ v16bf frag2(const __bf16* lo, const __bf16* hi) {
  v8bf a = *(const v8bf*)lo;
  v8bf b = *(const v8bf*)hi;
  return __builtin_shufflevector(a, b, 0,1,2,3,4,5,6,7,8,9,10,11,12,13,14,15);
}

static __device__ __forceinline__ v8f vzero8() {
  v8f z = {0.f,0.f,0.f,0.f,0.f,0.f,0.f,0.f};
  return z;
}

// ---------------------------------------------------------------------------
// elementwise f32 -> bf16 (weight layouts already match the [N][K] B-operand)
__global__ void k_cvt(const float* __restrict__ src, __bf16* __restrict__ dst, int n) {
  int i = blockIdx.x * blockDim.x + threadIdx.x;
  if (i < n) dst[i] = (__bf16)src[i];
}

// x [B,C,P] f32 -> xT [B,P,C] bf16   (A-operand of Q-proj GEMM)
__global__ void k_xT(const float* __restrict__ x, __bf16* __restrict__ xT) {
  int i  = blockIdx.x * blockDim.x + threadIdx.x;   // c fastest
  int c  = i & (C_ - 1);
  int bp = i >> 8;
  int p  = bp % P_;
  int b  = bp / P_;
  xT[i] = (__bf16)x[((size_t)(b * C_ + c)) * P_ + p];
}

// im2col for the 7x7/stride-7 conv: patches[b][t][k], k = c*49 + r*7 + s
__global__ void k_im2col(const float* __restrict__ x, __bf16* __restrict__ patches) {
  int i  = blockIdx.x * blockDim.x + threadIdx.x;   // k fastest (k range = 12544)
  int k  = i % P_;
  int bt = i / P_;
  int t  = bt & 255;
  int b  = bt >> 8;
  int c  = k / 49;
  int rs = k % 49;
  int r  = rs / 7, s = rs % 7;
  int ty = t >> 4, tx = t & 15;
  int yy = ty * 7 + r, xx = tx * 7 + s;
  patches[i] = (__bf16)x[(((size_t)b * C_ + c) * HW_ + yy) * HW_ + xx];
}

// ---------------------------------------------------------------------------
// GEMM core: block = 256 thr (8 waves) computes a 256x64 tile of
// C[M,N] = A[M,K] * B[N,K]^T. Each wave owns a 32x64 strip (2 M-tiles x
// 4 N-tiles), so each B fragment feeds two WMMAs. K staged in 32-wide steps
// through double-buffered LDS with async global->LDS copies.
static __device__ __forceinline__ void gemm_tiles(const __bf16* __restrict__ A,
                                                  const __bf16* __restrict__ Bm,
                                                  int K, int m0, int n0,
                                                  v8f acc[2][4]) {
  __shared__ __bf16 As[2][256 * 32];   // 2 x 16 KB
  __shared__ __bf16 Bs[2][64 * 32];    // 2 x  4 KB
  const int tid = threadIdx.x;
  const int wv = tid >> 5, lane = tid & 31;
  const int lg = lane >> 4, ln = lane & 15;

  auto issue = [&](int bb, int k0) {
    for (int it = 0; it < 4; ++it) {           // A tile: 1024 x 16B chunks
      int ch = tid + 256 * it;
      int row = ch >> 2, off = ch & 3;
      cp16(&As[bb][row * 32 + off * 8],
           A + (size_t)(m0 + row) * K + k0 + off * 8);
    }
    {                                          // B tile: 256 x 16B chunks
      int row = tid >> 2, off = tid & 3;
      cp16(&Bs[bb][row * 32 + off * 8],
           Bm + (size_t)(n0 + row) * K + k0 + off * 8);
    }
  };

  issue(0, 0);
  wait_async();
  __syncthreads();

  const int nk = K / 32;
  for (int i = 0; i < nk; ++i) {
    const int bb = i & 1;
    if (i + 1 < nk) issue(bb ^ 1, (i + 1) * 32);

    const __bf16* ar0 = &As[bb][(wv * 32 + ln) * 32];
    const __bf16* ar1 = ar0 + 16 * 32;
    v16bf a0 = frag2(ar0 + lg * 8, ar0 + 16 + lg * 8);
    v16bf a1 = frag2(ar1 + lg * 8, ar1 + 16 + lg * 8);
    for (int n = 0; n < 4; ++n) {
      const __bf16* br = &Bs[bb][(n * 16 + ln) * 32 + lg * 8];
      v16bf bf = frag2(br, br + 16);
      acc[0][n] = wmma_bf16(a0, bf, acc[0][n]);
      acc[1][n] = wmma_bf16(a1, bf, acc[1][n]);
    }
    wait_async();
    __syncthreads();
  }
}

// bf16-out GEMM
__global__ __launch_bounds__(256) void k_gemm(const __bf16* __restrict__ A,
                                              const __bf16* __restrict__ Bm,
                                              __bf16* __restrict__ Cm,
                                              int N, int K) {
  v8f acc[2][4];
  for (int m = 0; m < 2; ++m)
    for (int n = 0; n < 4; ++n) acc[m][n] = vzero8();
  const int m0 = blockIdx.x * 256, n0 = blockIdx.y * 64;
  gemm_tiles(A, Bm, K, m0, n0, acc);

  const int tid = threadIdx.x;
  const int wv = tid >> 5, lane = tid & 31;
  const int lg = lane >> 4, ln = lane & 15;
  for (int mt = 0; mt < 2; ++mt)
    for (int n = 0; n < 4; ++n)
      for (int r = 0; r < 8; ++r) {
        int mm = m0 + wv * 32 + mt * 16 + r + 8 * lg;
        int nn = n0 + n * 16 + ln;
        Cm[(size_t)mm * N + nn] = (__bf16)acc[mt][n][r];
      }
}

// Out-proj GEMM variant: +bias, f32 output scattered to NCHW [b][oc][p]
__global__ __launch_bounds__(256) void k_gemm_out(const __bf16* __restrict__ A,
                                                  const __bf16* __restrict__ Bm,
                                                  const float* __restrict__ bias,
                                                  float* __restrict__ out,
                                                  int N, int K) {
  v8f acc[2][4];
  for (int m = 0; m < 2; ++m)
    for (int n = 0; n < 4; ++n) acc[m][n] = vzero8();
  const int m0 = blockIdx.x * 256, n0 = blockIdx.y * 64;
  gemm_tiles(A, Bm, K, m0, n0, acc);

  const int tid = threadIdx.x;
  const int wv = tid >> 5, lane = tid & 31;
  const int lg = lane >> 4, ln = lane & 15;
  for (int n = 0; n < 4; ++n) {
    int oc = n0 + n * 16 + ln;
    float bv = bias[oc];
    for (int mt = 0; mt < 2; ++mt)
      for (int r = 0; r < 8; ++r) {
        int mm = m0 + wv * 32 + mt * 16 + r + 8 * lg;
        int b = mm / P_, p = mm % P_;
        out[((size_t)b * C_ + oc) * P_ + p] = acc[mt][n][r] + bv;
      }
  }
}

// ---------------------------------------------------------------------------
// Fused attention: per block = one (b,head) x 128 query rows (8 waves x 16).
// K[256x64] staged in LDS (async) for QK^T; same 32KB buffer then reused as
// V^T[64x256] for PV. Scores live in the WMMA accumulator layout; softmax via
// half-wave shfl_xor reductions; P restaged through 1KB/wave LDS.
__global__ __launch_bounds__(256) void k_attn(const __bf16* __restrict__ qtok,
                                              const __bf16* __restrict__ kvtok,
                                              __bf16* __restrict__ otok) {
  __shared__ __bf16 kvbuf[NKV_ * 64];   // 32 KB: K[j][dd], then V^T[dd][j]
  __shared__ __bf16 pbuf[8 * 16 * 32];  // 8 KB:  per-wave P staging
  const int tid = threadIdx.x;
  const int wv = tid >> 5, lane = tid & 31;
  const int lg = lane >> 4, ln = lane & 15;
  const int bh = blockIdx.y;
  const int b = bh >> 3, hd = bh & 7;
  const int p0 = blockIdx.x * 128;

  // ---- stage K[j][dd] (row = 64 bf16), async where available ----
  for (int it = 0; it < 8; ++it) {
    int ch = tid + 256 * it;
    int j = ch >> 3, off = (ch & 7) * 8;
    cp16(&kvbuf[j * 64 + off],
         kvtok + (size_t)(b * NKV_ + j) * 1024 + hd * 64 + off);
  }
  wait_async();
  __syncthreads();

  // ---- Q A-fragments straight from global (rows are contiguous 64 elems) ----
  const __bf16* qr = qtok + (size_t)(b * P_ + p0 + wv * 16 + ln) * INNER_ + hd * 64;
  v16bf aq0 = frag2(qr + lg * 8,       qr + 16 + lg * 8);
  v16bf aq1 = frag2(qr + 32 + lg * 8,  qr + 48 + lg * 8);

  // ---- scores S[16 rows][256 cols] as 16 accumulator tiles ----
  v8f S[16];
  for (int t = 0; t < 16; ++t) {
    v8f a = vzero8();
    const __bf16* kr = &kvbuf[(t * 16 + ln) * 64];
    a = wmma_bf16(aq0, frag2(kr + lg * 8,      kr + 16 + lg * 8), a);
    a = wmma_bf16(aq1, frag2(kr + 32 + lg * 8, kr + 48 + lg * 8), a);
    S[t] = a;
  }

  // ---- softmax over j (per row). Row r+8*lg lives on the 16 lanes sharing lg,
  //      one col per lane per tile -> lane-local reduce over t, then xor 1..8.
  for (int r = 0; r < 8; ++r) {
    float m = S[0][r];
    for (int t = 1; t < 16; ++t) m = fmaxf(m, S[t][r]);
    for (int off = 1; off < 16; off <<= 1) m = fmaxf(m, __shfl_xor(m, off, 32));
    float sum = 0.f;
    for (int t = 0; t < 16; ++t) {
      float e = __expf((S[t][r] - m) * SCALE_);
      S[t][r] = e;
      sum += e;
    }
    for (int off = 1; off < 16; off <<= 1) sum += __shfl_xor(sum, off, 32);
    float inv = 1.0f / sum;
    for (int t = 0; t < 16; ++t) S[t][r] *= inv;
  }

  __syncthreads();   // everyone done reading K region

  // ---- stage V^T[dd][j] into the same buffer (transpose on LDS write) ----
  for (int it = 0; it < 8; ++it) {
    int ch = tid + 256 * it;
    int j = ch >> 3, dq = (ch & 7) * 8;
    union { uint4 u; __bf16 h[8]; } uv;
    uv.u = *(const uint4*)(kvtok + (size_t)(b * NKV_ + j) * 1024 + 512 + hd * 64 + dq);
    for (int e = 0; e < 8; ++e) kvbuf[(dq + e) * NKV_ + j] = uv.h[e];
  }
  __syncthreads();

  // ---- O = P * V, streaming P in 32-wide j chunks through per-wave LDS ----
  v8f O[4];
  for (int n = 0; n < 4; ++n) O[n] = vzero8();
  __bf16* myp = &pbuf[wv * 512];          // private 16x32 staging
  for (int c = 0; c < 8; ++c) {
    for (int tt = 0; tt < 2; ++tt) {
      int t = 2 * c + tt;
      for (int r = 0; r < 8; ++r)
        myp[(r + 8 * lg) * 32 + tt * 16 + ln] = (__bf16)S[t][r];
    }
    asm volatile("" ::: "memory");        // same-wave DS ops are in-order
    const __bf16* pr = &myp[ln * 32];
    v16bf ap = frag2(pr + lg * 8, pr + 16 + lg * 8);
    for (int n = 0; n < 4; ++n) {
      const __bf16* vr = &kvbuf[(n * 16 + ln) * NKV_ + c * 32 + lg * 8];
      O[n] = wmma_bf16(ap, frag2(vr, vr + 16), O[n]);
    }
    asm volatile("" ::: "memory");
  }

  // ---- write out tokens [b][p][h*64+dd] (A-operand of out-proj) ----
  for (int n = 0; n < 4; ++n)
    for (int r = 0; r < 8; ++r)
      otok[(size_t)(b * P_ + p0 + wv * 16 + r + 8 * lg) * INNER_ +
           hd * 64 + n * 16 + ln] = (__bf16)O[n][r];
}

// ---------------------------------------------------------------------------
extern "C" void kernel_launch(void* const* d_in, const int* in_sizes, int n_in,
                              void* d_out, int out_size, void* d_ws, size_t ws_size,
                              hipStream_t stream) {
  const float* x    = (const float*)d_in[0];
  const float* Wq   = (const float*)d_in[1];
  const float* Wkv  = (const float*)d_in[2];
  const float* Wout = (const float*)d_in[3];
  const float* bout = (const float*)d_in[4];
  float* out = (float*)d_out;

  // workspace layout (bytes); otok aliases xT+patches (both dead by then)
  char* ws = (char*)d_ws;
  __bf16* xT      = (__bf16*)(ws + 0);           // 25,690,112 B
  __bf16* patches = (__bf16*)(ws + 25690112);    // 25,690,112 B
  __bf16* otok    = (__bf16*)(ws + 0);           // 51,380,224 B (alias)
  __bf16* qtok    = (__bf16*)(ws + 51380224);    // 51,380,224 B
  __bf16* Wq_b    = (__bf16*)(ws + 102760448);   //    262,144 B
  __bf16* Wout_b  = (__bf16*)(ws + 103022592);   //    262,144 B
  __bf16* Wkv_b   = (__bf16*)(ws + 103284736);   // 25,690,112 B
  __bf16* kvtok   = (__bf16*)(ws + 128974848);   //  2,097,152 B  (total ~131 MB)

  const int nx = B_ * C_ * P_;                   // 12,845,056 (divisible by 256)
  k_xT    <<<nx / 256, 256, 0, stream>>>(x, xT);
  k_im2col<<<nx / 256, 256, 0, stream>>>(x, patches);   // B*256*12544 == nx
  k_cvt<<<(512 * 256) / 256, 256, 0, stream>>>(Wq, Wq_b, 512 * 256);
  k_cvt<<<(1024 * 12544) / 256, 256, 0, stream>>>(Wkv, Wkv_b, 1024 * 12544);
  k_cvt<<<(256 * 512) / 256, 256, 0, stream>>>(Wout, Wout_b, 256 * 512);

  dim3 gq(50176 / 256, 512 / 64);                // Q proj: 196 x 8
  k_gemm<<<gq, 256, 0, stream>>>(xT, Wq_b, qtok, 512, 256);

  dim3 gkv(1024 / 256, 1024 / 64);               // KV conv GEMM: 4 x 16
  k_gemm<<<gkv, 256, 0, stream>>>(patches, Wkv_b, kvtok, 1024, 12544);

  dim3 ga(P_ / 128, B_ * HEADS_);                // fused attention: 98 x 32
  k_attn<<<ga, 256, 0, stream>>>(qtok, kvtok, otok);

  dim3 go(50176 / 256, 256 / 64);                // out proj: 196 x 4
  k_gemm_out<<<go, 256, 0, stream>>>(otok, Wout_b, bout, out, 256, 512);
}
